// RETAIN_29489245454839
// MI455X (gfx1250) — matrix operand order repaired and checked by reading, not compiled, not run on hardware
//
#include <hip/hip_runtime.h>
#include <math.h>

// ---------------- problem constants ----------------
#define BB    128
#define SEQL  200
#define VOCAB 1400
#define VPAD  1408     // VOCAB padded to multiple of 32 (zero-filled)
#define DD    256
#define HH    256
#define G4H   1024     // 4*HH
#define NCLS  128

typedef __attribute__((ext_vector_type(16))) _Float16 v16h;
typedef __attribute__((ext_vector_type(8)))  float    v8f;

union AF { v16h v; _Float16 h[16]; float4 q[2]; };
union CF { v8f  v; float f[8]; };

__device__ __forceinline__ v8f wmma_f16(v16h a, v16h b, v8f c) {
  // D = A(16x32 f16) * B(32x16 f16) + C(16x16 f32)
  return __builtin_amdgcn_wmma_f32_16x16x32_f16(false, a, false, b, (short)0, c, false, false);
}

// ---- fast transcendentals: CDNA5 has native v_tanh_f32 (TRANS32, co-executes with WMMA) ----
#if __has_builtin(__builtin_amdgcn_tanhf)
__device__ __forceinline__ float fast_tanh(float x) { return __builtin_amdgcn_tanhf(x); }
#elif __has_builtin(__builtin_amdgcn_tanh_f32)
__device__ __forceinline__ float fast_tanh(float x) { return __builtin_amdgcn_tanh_f32(x); }
#else
__device__ __forceinline__ float fast_tanh(float x) { return tanhf(x); }
#endif

// sigmoid(x) = 0.5 + 0.5*tanh(x/2): 3 VALU ops with native tanh
__device__ __forceinline__ float sigm(float x) {
  return 0.5f + 0.5f * fast_tanh(0.5f * x);
}

// A fragment (16x32, f16) from row-major f16 (LDS or global), row stride `ld` halves.
// ISA layout: lane m=lane&15; halves 0..7 -> K=ko..ko+7, halves 8..15 -> K=ko+16..ko+23, ko=8*(lane>=16)
__device__ __forceinline__ v16h frag_a(const _Float16* base, int ld, int lane) {
  AF a;
  const _Float16* p = base + (lane & 15) * ld + ((lane >> 4) << 3);
  a.q[0] = *(const float4*)(p);
  a.q[1] = *(const float4*)(p + 16);
  return a.v;
}

// B fragment (32x16, f16) taken as W^T where W is row-major [N,K] f16.
// ISA layout: lane n=lane&15; halves 0..7 -> K=k0+ko..+7, halves 8..15 -> K=k0+ko+16..+23, ko=8*(lane>=16)
__device__ __forceinline__ v16h frag_b(const _Float16* W, int ldw, int nrow, int k0, int lane) {
  AF a;
  const _Float16* p = W + (size_t)(nrow + (lane & 15)) * ldw + k0 + ((lane >> 4) << 3);
  a.q[0] = *(const float4*)(p);
  a.q[1] = *(const float4*)(p + 16);
  return a.v;
}

// ---------------- prep kernels ----------------
__global__ void k_cvt_f16(const float* __restrict__ s, _Float16* __restrict__ d, int n) {
  int i = blockIdx.x * blockDim.x + threadIdx.x;
  if (i < n) d[i] = (_Float16)s[i];
}

// embT[d, v] = emb[v, d]  (f16, K padded/zeroed to VPAD)
__global__ void k_emb_T(const float* __restrict__ emb, _Float16* __restrict__ embT) {
  int i = blockIdx.x * blockDim.x + threadIdx.x;       // over DD*VPAD
  if (i >= DD * VPAD) return;
  int dcol = i / VPAD, v = i - dcol * VPAD;
  _Float16 val = (_Float16)0.0f;
  if (v < VOCAB) val = (_Float16)emb[(size_t)v * DD + dcol];
  embT[i] = val;
}

// ---------------- generic WMMA GEMM: C[M,N] = act(A[M,K](f32) * W[N,K]^T(f16) + bias) ----------------
// tile: 128(M) x 64(N), 256 threads = 8 waves, each wave a 32x32 quadrant.
__global__ __launch_bounds__(256)
void k_gemm(const float* __restrict__ A, int lda,
            const _Float16* __restrict__ W, int ldw,
            const float* __restrict__ bias,
            float* __restrict__ Cf, _Float16* __restrict__ Ch, int ldc,
            int K, int act) {
  __shared__ _Float16 As[128 * 32];                    // 8 KB, one K-chunk of A in f16
  const int tid = threadIdx.x, w = tid >> 5, lane = tid & 31;
  const int hi8 = (lane >> 4) << 3;
  const int m_blk = blockIdx.y * 128, n_blk = blockIdx.x * 64;
  const int m_off = (w >> 1) << 5, n_off = (w & 1) << 5;

  CF acc[2][2];
  #pragma unroll
  for (int i = 0; i < 2; i++)
    #pragma unroll
    for (int j = 0; j < 2; j++)
      #pragma unroll
      for (int r = 0; r < 8; r++) acc[i][j].f[r] = 0.0f;

  const int srow = tid >> 1, skh = (tid & 1) << 4;     // staging assignment: 2 thr/row, 16 elems
  const float* __restrict__ Arow = A + (size_t)(m_blk + srow) * lda;

  const int nch = (K + 31) >> 5;
  for (int kc = 0; kc < nch; kc++) {
    const int k0 = kc << 5;
    __syncthreads();
    {   // stage A chunk: 128 rows x 32 halves, branch-free
      _Float16 tmp[16];
      if (k0 + 32 <= K) {                              // uniform branch: full chunk, vector loads
        const float4* s4 = (const float4*)(Arow + k0 + skh);
        float4 f[4];
        #pragma unroll
        for (int i = 0; i < 4; i++) f[i] = s4[i];
        const float* fs = (const float*)f;
        #pragma unroll
        for (int i = 0; i < 16; i++) tmp[i] = (_Float16)fs[i];
      } else {                                         // tail chunk: clamped unconditional loads
        #pragma unroll
        for (int i = 0; i < 16; i++) {
          int k  = k0 + skh + i;
          int kk = (k < K) ? k : (K - 1);              // always in-bounds address
          float v = Arow[kk];                          // unconditional global_load_b32
          tmp[i] = (k < K) ? (_Float16)v : (_Float16)0.0f;  // v_cndmask, no exec branch
        }
      }
      *(float4*)&As[srow * 32 + skh]     = *(float4*)&tmp[0];
      *(float4*)&As[srow * 32 + skh + 8] = *(float4*)&tmp[8];
    }
    __syncthreads();
    v16h a0 = frag_a(As + (m_off)      * 32, 32, lane);
    v16h a1 = frag_a(As + (m_off + 16) * 32, 32, lane);
    v16h b0 = frag_b(W, ldw, n_blk + n_off,      k0, lane);
    v16h b1 = frag_b(W, ldw, n_blk + n_off + 16, k0, lane);
    acc[0][0].v = wmma_f16(a0, b0, acc[0][0].v);
    acc[0][1].v = wmma_f16(a0, b1, acc[0][1].v);
    acc[1][0].v = wmma_f16(a1, b0, acc[1][0].v);
    acc[1][1].v = wmma_f16(a1, b1, acc[1][1].v);
  }

  #pragma unroll
  for (int i = 0; i < 2; i++)
    #pragma unroll
    for (int j = 0; j < 2; j++) {
      int n = n_blk + n_off + (j << 4) + (lane & 15);
      float bv = bias ? bias[n] : 0.0f;
      #pragma unroll
      for (int r = 0; r < 8; r++) {
        int m = m_blk + m_off + (i << 4) + hi8 + r;
        float v = acc[i][j].f[r] + bv;
        if (act == 1) v = fast_tanh(v);
        if (Ch) Ch[(size_t)m * ldc + n] = (_Float16)v;
        else    Cf[(size_t)m * ldc + n] = v;
      }
    }
}

// ---------------- persistent TimeLSTM scan ----------------
// grid = (4 batch-groups of 32 rows, 2 LSTMs); 256 threads = 8 waves.
// wave w owns hidden cols [w*32, w*32+32): 2 n-tiles x {2 m-tiles} x {4 gates}.
struct LstmP {
  const _Float16* ux;    // [B, SEQ, 4H] f16 (includes Uall bias)
  const _Float16* Wall;  // [4H, H] f16
  const float*    ball;  // [4H]
  const _Float16* Wd;    // [H, H] f16
  const float*    bd;    // [H]
  const float*    h0;    // [B, H]
  const float*    c0;    // [B, H]
  float*          out;   // [B, SEQ, H]
};

__global__ __launch_bounds__(256)
void k_scan(LstmP p1, LstmP p2, const float* __restrict__ ts) {
  LstmP P = (blockIdx.y == 0) ? p1 : p2;
  const int b0 = blockIdx.x * 32;                       // this WG's batch slice
  __shared__ _Float16 hbuf[32 * HH];                    // 16 KB f16 state for WMMA A
  __shared__ _Float16 cbuf[32 * HH];                    // 16 KB
  const int tid = threadIdx.x, w = tid >> 5, lane = tid & 31;
  const int rl = lane & 15, hi8 = (lane >> 4) << 3;
  const int nc0 = w << 5;                               // wave's hidden-col base

  for (int i = tid; i < 32 * HH; i += 256) {            // init LDS state
    int row = i >> 8, col = i & 255;
    hbuf[i] = (_Float16)P.h0[(size_t)(b0 + row) * HH + col];
    cbuf[i] = (_Float16)P.c0[(size_t)(b0 + row) * HH + col];
  }
  CF cacc[2][2];                                        // fp32 master cell state, D-layout
  float ballv[4][2], bdv[2];
  #pragma unroll
  for (int nt = 0; nt < 2; nt++) {
    int n = nc0 + (nt << 4) + rl;
    bdv[nt] = P.bd[n];
    #pragma unroll
    for (int g = 0; g < 4; g++) ballv[g][nt] = P.ball[g * HH + n];
    #pragma unroll
    for (int mt = 0; mt < 2; mt++)
      #pragma unroll
      for (int r = 0; r < 8; r++)
        cacc[mt][nt].f[r] = P.c0[(size_t)(b0 + (mt << 4) + hi8 + r) * HH + n];
  }
  __syncthreads();

  for (int t = 0; t < SEQL; t++) {
    if (t + 1 < SEQL)                                   // hint next step's ux into caches
      __builtin_prefetch(P.ux + ((size_t)b0 * SEQL + (t + 1)) * G4H, 0, 1);

    // ---- c_s1 = tanh(c @ Wd^T + bd) ----
    CF sac[2][2];
    #pragma unroll
    for (int mt = 0; mt < 2; mt++)
      #pragma unroll
      for (int nt = 0; nt < 2; nt++)
        #pragma unroll
        for (int r = 0; r < 8; r++) sac[mt][nt].f[r] = 0.0f;
    #pragma unroll
    for (int kc = 0; kc < 8; kc++) {
      int k0 = kc << 5;
      v16h a0 = frag_a(cbuf + k0,           HH, lane);
      v16h a1 = frag_a(cbuf + 16 * HH + k0, HH, lane);
      #pragma unroll
      for (int nt = 0; nt < 2; nt++) {
        v16h bf = frag_b(P.Wd, HH, nc0 + (nt << 4), k0, lane);
        sac[0][nt].v = wmma_f16(a0, bf, sac[0][nt].v);
        sac[1][nt].v = wmma_f16(a1, bf, sac[1][nt].v);
      }
    }
    CF cadj[2][2];
    #pragma unroll
    for (int mt = 0; mt < 2; mt++)
      #pragma unroll
      for (int nt = 0; nt < 2; nt++)
        #pragma unroll
        for (int r = 0; r < 8; r++) {
          float s  = fast_tanh(sac[mt][nt].f[r] + bdv[nt]);
          float tv = ts[(size_t)(b0 + (mt << 4) + hi8 + r) * SEQL + t];
          cadj[mt][nt].f[r] = cacc[mt][nt].f[r] - s + s * tv;
        }

    // ---- gates = sigmoid(h @ Wall^T + ball + ux_t) ----
    CF gac[2][2][4];
    #pragma unroll
    for (int mt = 0; mt < 2; mt++)
      #pragma unroll
      for (int nt = 0; nt < 2; nt++)
        #pragma unroll
        for (int g = 0; g < 4; g++)
          #pragma unroll
          for (int r = 0; r < 8; r++) gac[mt][nt][g].f[r] = 0.0f;
    #pragma unroll 2
    for (int kc = 0; kc < 8; kc++) {
      int k0 = kc << 5;
      v16h a0 = frag_a(hbuf + k0,           HH, lane);
      v16h a1 = frag_a(hbuf + 16 * HH + k0, HH, lane);
      #pragma unroll
      for (int g = 0; g < 4; g++)
        #pragma unroll
        for (int nt = 0; nt < 2; nt++) {
          v16h bf = frag_b(P.Wall, HH, g * HH + nc0 + (nt << 4), k0, lane);
          gac[0][nt][g].v = wmma_f16(a0, bf, gac[0][nt][g].v);
          gac[1][nt][g].v = wmma_f16(a1, bf, gac[1][nt][g].v);
        }
    }
    __syncthreads();                                    // all LDS reads for step t done

    // ---- pointwise update, write h/c back ----
    #pragma unroll
    for (int mt = 0; mt < 2; mt++)
      #pragma unroll
      for (int nt = 0; nt < 2; nt++) {
        int n = nc0 + (nt << 4) + rl;
        #pragma unroll
        for (int r = 0; r < 8; r++) {
          int bl = (mt << 4) + hi8 + r;
          int b  = b0 + bl;
          const _Float16* uxp = P.ux + ((size_t)b * SEQL + t) * G4H;
          float fg = sigm(gac[mt][nt][0].f[r] + ballv[0][nt] + (float)uxp[0 * HH + n]);
          float ig = sigm(gac[mt][nt][1].f[r] + ballv[1][nt] + (float)uxp[1 * HH + n]);
          float og = sigm(gac[mt][nt][2].f[r] + ballv[2][nt] + (float)uxp[2 * HH + n]);
          float cg = sigm(gac[mt][nt][3].f[r] + ballv[3][nt] + (float)uxp[3 * HH + n]);
          float cn = fg * cadj[mt][nt].f[r] + ig * cg;
          float hn = og * fast_tanh(cn);
          cacc[mt][nt].f[r] = cn;
          hbuf[bl * HH + n] = (_Float16)hn;
          cbuf[bl * HH + n] = (_Float16)cn;
          P.out[((size_t)b * SEQL + t) * HH + n] = hn;
        }
      }
    __syncthreads();                                    // state visible for step t+1
  }
}

// ---------------- attention kernels ----------------
__global__ void k_e(const float* __restrict__ out1, const float* __restrict__ wa,
                    float* __restrict__ E) {
  int wid = (blockIdx.x * blockDim.x + threadIdx.x) >> 5;
  int lane = threadIdx.x & 31;
  if (wid >= BB * SEQL) return;
  const float* row = out1 + (size_t)wid * HH;
  float s = 0.0f;
  for (int i = lane; i < HH; i += 32) s += row[i] * wa[i];
  #pragma unroll
  for (int off = 16; off; off >>= 1) s += __shfl_xor(s, off, 32);
  if (lane == 0) E[wid] = s;
}

__global__ void k_softmax(const float* __restrict__ E, float* __restrict__ alpha) {
  __shared__ float red[256];
  int b = blockIdx.x, s = threadIdx.x;
  float v = (s < SEQL) ? E[(size_t)b * SEQL + s] : -3.0e38f;
  red[s] = v; __syncthreads();
  for (int off = 128; off; off >>= 1) {
    if (s < off) red[s] = fmaxf(red[s], red[s + off]);
    __syncthreads();
  }
  float mx = red[0]; __syncthreads();
  float e = (s < SEQL) ? __expf(v - mx) : 0.0f;
  red[s] = e; __syncthreads();
  for (int off = 128; off; off >>= 1) {
    if (s < off) red[s] += red[s + off];
    __syncthreads();
  }
  if (s < SEQL) alpha[(size_t)b * SEQL + s] = e / red[0];
}

__global__ void k_ctx(const float* __restrict__ embedded, const float* __restrict__ Beta,
                      const float* __restrict__ alpha, float* __restrict__ ctx) {
  int idx = blockIdx.x * blockDim.x + threadIdx.x;     // BB*HH threads
  int b = idx >> 8, h = idx & 255;
  float acc = 0.0f;
  for (int s = 0; s < SEQL; s++) {
    size_t o = ((size_t)b * SEQL + s) * HH + h;
    acc += embedded[o] * Beta[o] * alpha[(size_t)b * SEQL + s];
  }
  ctx[(size_t)b * HH + h] = acc;
}

// ---------------- host orchestration ----------------
extern "C" void kernel_launch(void* const* d_in, const int* in_sizes, int n_in,
                              void* d_out, int out_size, void* d_ws, size_t ws_size,
                              hipStream_t stream) {
  const float* inputs  = (const float*)d_in[0];
  const float* tstamps = (const float*)d_in[1];
  const float* emb     = (const float*)d_in[2];
  const float* Wall1_w = (const float*)d_in[3];
  const float* Wall1_b = (const float*)d_in[4];
  const float* Uall1_w = (const float*)d_in[5];
  const float* Uall1_b = (const float*)d_in[6];
  const float* Wd1_w   = (const float*)d_in[7];
  const float* Wd1_b   = (const float*)d_in[8];
  const float* Wall2_w = (const float*)d_in[9];
  const float* Wall2_b = (const float*)d_in[10];
  const float* Uall2_w = (const float*)d_in[11];
  const float* Uall2_b = (const float*)d_in[12];
  const float* Wd2_w   = (const float*)d_in[13];
  const float* Wd2_b   = (const float*)d_in[14];
  const float* wa_w    = (const float*)d_in[15];
  const float* Wb_w    = (const float*)d_in[16];
  const float* Wout_w  = (const float*)d_in[17];
  const float* h01     = (const float*)d_in[18];
  const float* c01     = (const float*)d_in[19];
  const float* h02     = (const float*)d_in[20];
  const float* c02     = (const float*)d_in[21];

  // workspace carve (~213 MB total)
  size_t off = 0;
  char* base = (char*)d_ws;
  auto carve = [&](size_t bytes) -> void* {
    void* r = base + off; off += (bytes + 255) & ~(size_t)255; return r;
  };
  float*    embedded = (float*)   carve(sizeof(float)    * (size_t)BB * SEQL * DD);
  _Float16* embT     = (_Float16*)carve(sizeof(_Float16) * (size_t)DD * VPAD);
  _Float16* wu1      = (_Float16*)carve(sizeof(_Float16) * (size_t)G4H * DD);
  _Float16* wu2      = (_Float16*)carve(sizeof(_Float16) * (size_t)G4H * DD);
  _Float16* wall1    = (_Float16*)carve(sizeof(_Float16) * (size_t)G4H * HH);
  _Float16* wall2    = (_Float16*)carve(sizeof(_Float16) * (size_t)G4H * HH);
  _Float16* wd1      = (_Float16*)carve(sizeof(_Float16) * (size_t)HH * HH);
  _Float16* wd2      = (_Float16*)carve(sizeof(_Float16) * (size_t)HH * HH);
  _Float16* wb16     = (_Float16*)carve(sizeof(_Float16) * (size_t)HH * HH);
  _Float16* wout16   = (_Float16*)carve(sizeof(_Float16) * (size_t)NCLS * HH);
  _Float16* ux1      = (_Float16*)carve(sizeof(_Float16) * (size_t)BB * SEQL * G4H);
  _Float16* ux2      = (_Float16*)carve(sizeof(_Float16) * (size_t)BB * SEQL * G4H);
  float*    out1     = (float*)   carve(sizeof(float)    * (size_t)BB * SEQL * HH);
  float*    out2     = (float*)   carve(sizeof(float)    * (size_t)BB * SEQL * HH);
  float*    Ebuf     = (float*)   carve(sizeof(float)    * (size_t)BB * SEQL);
  float*    alpha    = (float*)   carve(sizeof(float)    * (size_t)BB * SEQL);
  float*    Beta     = (float*)   carve(sizeof(float)    * (size_t)BB * SEQL * HH);
  float*    ctx      = (float*)   carve(sizeof(float)    * (size_t)BB * HH);

  // weight converts (fp32 -> f16, layout unchanged: torch [N,K])
  auto cvt = [&](const float* s, _Float16* d, int n) {
    k_cvt_f16<<<(n + 255) / 256, 256, 0, stream>>>(s, d, n);
  };
  cvt(Uall1_w, wu1,   G4H * DD);
  cvt(Uall2_w, wu2,   G4H * DD);
  cvt(Wall1_w, wall1, G4H * HH);
  cvt(Wall2_w, wall2, G4H * HH);
  cvt(Wd1_w,   wd1,   HH * HH);
  cvt(Wd2_w,   wd2,   HH * HH);
  cvt(Wb_w,    wb16,  HH * HH);
  cvt(Wout_w,  wout16, NCLS * HH);
  k_emb_T<<<(DD * VPAD + 255) / 256, 256, 0, stream>>>(emb, embT);

  const int MROWS = BB * SEQL;   // 25600
  // embedded = inputs @ emb          [25600,1400] x [1400,256]
  k_gemm<<<dim3(DD / 64, MROWS / 128), 256, 0, stream>>>(
      inputs, VOCAB, embT, VPAD, nullptr, embedded, nullptr, DD, VOCAB, 0);
  // ux = embedded @ Uall^T + Uall_b  -> f16 [B,SEQ,4H]
  k_gemm<<<dim3(G4H / 64, MROWS / 128), 256, 0, stream>>>(
      embedded, DD, wu1, DD, Uall1_b, nullptr, ux1, G4H, DD, 0);
  k_gemm<<<dim3(G4H / 64, MROWS / 128), 256, 0, stream>>>(
      embedded, DD, wu2, DD, Uall2_b, nullptr, ux2, G4H, DD, 0);

  // sequential scans: 4 batch-groups x 2 LSTMs, persistent over 200 steps
  LstmP p1{ux1, wall1, Wall1_b, wd1, Wd1_b, h01, c01, out1};
  LstmP p2{ux2, wall2, Wall2_b, wd2, Wd2_b, h02, c02, out2};
  k_scan<<<dim3(BB / 32, 2), 256, 0, stream>>>(p1, p2, tstamps);

  // attention
  k_e<<<(BB * SEQL) / 8, 256, 0, stream>>>(out1, wa_w, Ebuf);
  k_softmax<<<BB, 256, 0, stream>>>(Ebuf, alpha);
  // Beta = tanh(out2 @ Wb^T)
  k_gemm<<<dim3(HH / 64, MROWS / 128), 256, 0, stream>>>(
      out2, HH, wb16, HH, nullptr, Beta, nullptr, HH, HH, 1);
  k_ctx<<<(BB * HH) / 256, 256, 0, stream>>>(embedded, Beta, alpha, ctx);
  // logits = ctx @ Wout^T  [128,256] x [256,128]
  k_gemm<<<dim3(NCLS / 64, BB / 128), 256, 0, stream>>>(
      ctx, HH, wout16, HH, nullptr, (float*)d_out, nullptr, NCLS, HH, 0);
}
